// FullHeteroGCN_7507602833968
// MI455X (gfx1250) — matrix operand order repaired and checked by reading, not compiled, not run on hardware
//
#include <hip/hip_runtime.h>

#define NFEAT 128
#define RPB 64        // rows of A/X per block
#define LSTRIDE 136   // padded LDS row stride in halfs (272B = 17*16B, keeps 16B align)

typedef __attribute__((ext_vector_type(16))) __bf16 v16bf;
typedef __attribute__((ext_vector_type(4)))  __bf16 v4bf;
typedef __attribute__((ext_vector_type(8)))  float  v8f;

__device__ __forceinline__ __bf16 f2bf(float x) { return (__bf16)x; }

// ---------------------------------------------------------------- zero ----
__global__ void k_zero_f32(float* __restrict__ p, long long n) {
    long long t = (long long)blockIdx.x * blockDim.x + threadIdx.x;
    long long stride = (long long)gridDim.x * blockDim.x;
    for (; t < n; t += stride) p[t] = 0.0f;
}

// ------------------------------------------------- scatter mean (accum) ----
// 32 threads (one wave) per edge, 4 floats each (float4 load, 4x atomicAdd).
__global__ void k_scatter_accum(const float* __restrict__ xsrc,
                                const int* __restrict__ src,
                                const int* __restrict__ dst,
                                float* __restrict__ acc,
                                float* __restrict__ cnt, int E) {
    long long t = (long long)blockIdx.x * blockDim.x + threadIdx.x;
    int e = (int)(t >> 5);
    if (e >= E) return;
    int c4 = (int)(t & 31) * 4;
    int s = src[e], d = dst[e];
    const float4 v = *(const float4*)(xsrc + (long long)s * NFEAT + c4);
    float* a = acc + (long long)d * NFEAT + c4;
    atomicAdd(a + 0, v.x);
    atomicAdd(a + 1, v.y);
    atomicAdd(a + 2, v.z);
    atomicAdd(a + 3, v.w);
    if ((t & 31) == 0) atomicAdd(cnt + d, 1.0f);
}

// ------------------------------------------------------------- mean div ----
__global__ void k_mean_div(float* __restrict__ acc,
                           const float* __restrict__ cnt, long long nelem) {
    long long t = (long long)blockIdx.x * blockDim.x + threadIdx.x;
    if (t >= nelem) return;
    acc[t] = acc[t] / fmaxf(cnt[t >> 7], 1.0f);
}

// --------------------------------------------- fused SAGE GEMM via WMMA ----
// Out[M,128] (+)= alpha * (A @ Wl^T + bl + X @ Wr^T)   [optional ReLU]
// Block = 256 threads = 8 waves; block owns a 64-row stripe. Wave w owns
// output columns [16w,16w+16): its 8 B-fragments (K=256 fused, bf16) are
// loop-invariant -> cached in registers and reused across 4 row tiles.
// Steady state per tile: 8x2 ds_load_b128 (A-frags) + 8 v_wmma.
// ACC/RELU are compile-time so the epilogue is branch-free; partial blocks
// (only the last one per launch) take a guarded path.
template <int ACC, int RELU>
__global__ __launch_bounds__(256) void k_sage_gemm(
        const float* __restrict__ A, const float* __restrict__ X,
        const float* __restrict__ Wl, const float* __restrict__ bl,
        const float* __restrict__ Wr, float* __restrict__ Out,
        int M, float alpha) {
    __shared__ __bf16 sA[RPB * LSTRIDE];
    __shared__ __bf16 sX[RPB * LSTRIDE];

    const int tid = threadIdx.x;
    const int m0  = blockIdx.x * RPB;

    // Stage 64x128 stripes of A and X into LDS as bf16.
    // float4 global loads (b128) + 4-elem bf16 LDS stores (8B, aligned).
#pragma unroll
    for (int i = 0; i < (RPB * NFEAT) / (256 * 4); ++i) {
        int flat4 = tid + 256 * i;         // unit = 4 floats; 32 per row
        int row = flat4 >> 5, col4 = (flat4 & 31) * 4;
        int gr = m0 + row;
        float4 av = make_float4(0.f, 0.f, 0.f, 0.f), xv = av;
        if (gr < M) {
            av = *(const float4*)(A + (long long)gr * NFEAT + col4);
            xv = *(const float4*)(X + (long long)gr * NFEAT + col4);
        }
        v4bf ap, xp;
        ap[0] = f2bf(av.x); ap[1] = f2bf(av.y); ap[2] = f2bf(av.z); ap[3] = f2bf(av.w);
        xp[0] = f2bf(xv.x); xp[1] = f2bf(xv.y); xp[2] = f2bf(xv.z); xp[3] = f2bf(xv.w);
        *(v4bf*)(&sA[row * LSTRIDE + col4]) = ap;
        *(v4bf*)(&sX[row * LSTRIDE + col4]) = xp;
    }
    __syncthreads();

    const int lane = tid & 31;
    const int wave = tid >> 5;
    const int n0   = wave * 16;
    const int mrow = lane & 15;
    const int hi   = lane >> 4;

    // Register-resident B fragments: lane%16 = N col, elements = K (16-bit A/B table order).
    // B[k][n] = W[n*128 + k]  (Out = A @ W^T).
    v16bf bWl[4], bWr[4];
#pragma unroll
    for (int kk = 0; kk < 4; ++kk) {
        const int kb = kk * 32 + hi * 8;
#pragma unroll
        for (int j = 0; j < 16; ++j) {
            const int k = kb + (j < 8 ? j : j + 8);
            bWl[kk][j] = f2bf(Wl[(n0 + mrow) * NFEAT + k]);
            bWr[kk][j] = f2bf(Wr[(n0 + mrow) * NFEAT + k]);
        }
    }

    const int col = n0 + mrow;
    const float blv = bl[col];
    const bool full = (m0 + RPB) <= M;     // block-uniform

#pragma unroll
    for (int rt = 0; rt < RPB / 16; ++rt) {
        const int mbase = (rt * 16 + mrow) * LSTRIDE;

        v8f c = {0.f, 0.f, 0.f, 0.f, 0.f, 0.f, 0.f, 0.f};

        // --- A @ Wl^T ---
#pragma unroll
        for (int kk = 0; kk < 4; ++kk) {
            v16bf a;
            const int kc = kk * 32;
#pragma unroll
            for (int t = 0; t < 8; ++t) {   // ISA 16-bit A 16x32 layout
                a[t]     = sA[mbase + kc + hi * 8 + t];
                a[t + 8] = sA[mbase + kc + 16 + hi * 8 + t];
            }
            c = __builtin_amdgcn_wmma_f32_16x16x32_bf16(false, a, false, bWl[kk],
                                                        (short)0, c, false, false);
        }
        // --- X @ Wr^T ---
#pragma unroll
        for (int kk = 0; kk < 4; ++kk) {
            v16bf a;
            const int kc = kk * 32;
#pragma unroll
            for (int t = 0; t < 8; ++t) {
                a[t]     = sX[mbase + kc + hi * 8 + t];
                a[t + 8] = sX[mbase + kc + 16 + hi * 8 + t];
            }
            c = __builtin_amdgcn_wmma_f32_16x16x32_bf16(false, a, false, bWr[kk],
                                                        (short)0, c, false, false);
        }

        // Epilogue: C/D layout -> row = m0 + rt*16 + r + hi*8, col = n0 + lane%16.
        const long long rbase = (long long)(m0 + rt * 16 + hi * 8) * NFEAT + col;
        if (full) {
#pragma unroll
            for (int r = 0; r < 8; ++r) {
                long long o = rbase + (long long)r * NFEAT;
                float v = alpha * (c[r] + blv);
                if (ACC) v += Out[o];
                if (RELU) v = fmaxf(v, 0.0f);
                Out[o] = v;
            }
        } else {
#pragma unroll
            for (int r = 0; r < 8; ++r) {
                int row = m0 + rt * 16 + r + hi * 8;
                if (row < M) {
                    long long o = (long long)row * NFEAT + col;
                    float v = alpha * (c[r] + blv);
                    if (ACC) v += Out[o];
                    if (RELU) v = fmaxf(v, 0.0f);
                    Out[o] = v;
                }
            }
        }
    }
}

// ------------------------------------------------------------- head -------
// out[m,c] = p2[m,:] . lin_W[c,:] + lin_b[c]   (C = 8)
__global__ void k_head(const float* __restrict__ p2,
                       const float* __restrict__ W,
                       const float* __restrict__ b,
                       float* __restrict__ out, int M) {
    long long t = (long long)blockIdx.x * blockDim.x + threadIdx.x;
    if (t >= (long long)M * 8) return;
    int m = (int)(t >> 3), c = (int)(t & 7);
    const float* row = p2 + (long long)m * NFEAT;
    const float* w = W + c * NFEAT;
    float acc = b[c];
#pragma unroll 8
    for (int k = 0; k < NFEAT; ++k) acc += row[k] * w[k];
    out[t] = acc;
}

// --------------------------------------------------------------------------
extern "C" void kernel_launch(void* const* d_in, const int* in_sizes, int n_in,
                              void* d_out, int out_size, void* d_ws, size_t ws_size,
                              hipStream_t stream) {
    const float* x_p    = (const float*)d_in[0];
    const float* x_a    = (const float*)d_in[1];
    const int*   c_src  = (const int*)d_in[2];
    const int*   c_dst  = (const int*)d_in[3];
    const int*   ao_src = (const int*)d_in[4];
    const int*   ao_dst = (const int*)d_in[5];
    const int*   ha_src = (const int*)d_in[6];
    const int*   ha_dst = (const int*)d_in[7];
    const float* l1c_Wl = (const float*)d_in[8];
    const float* l1c_bl = (const float*)d_in[9];
    const float* l1c_Wr = (const float*)d_in[10];
    const float* l1a_Wl = (const float*)d_in[11];
    const float* l1a_bl = (const float*)d_in[12];
    const float* l1a_Wr = (const float*)d_in[13];
    const float* l1h_Wl = (const float*)d_in[14];
    const float* l1h_bl = (const float*)d_in[15];
    const float* l1h_Wr = (const float*)d_in[16];
    const float* l2c_Wl = (const float*)d_in[17];
    const float* l2c_bl = (const float*)d_in[18];
    const float* l2c_Wr = (const float*)d_in[19];
    const float* l2a_Wl = (const float*)d_in[20];
    const float* l2a_bl = (const float*)d_in[21];
    const float* l2a_Wr = (const float*)d_in[22];
    // d_in[23..25] = l2h_* : dead code in the reference.
    const float* lin_W  = (const float*)d_in[26];
    const float* lin_b  = (const float*)d_in[27];

    const int NP = in_sizes[0] / NFEAT;
    const int NA = in_sizes[1] / NFEAT;
    const int Ec = in_sizes[2], Ea = in_sizes[4], Eh = in_sizes[6];

    float* ws = (float*)d_ws;
    const long long NPf = (long long)NP * NFEAT;
    const long long NAf = (long long)NA * NFEAT;
    float* accC = ws;                       // [NP,128] cites mean accum
    float* accA = ws + NPf;                 // [NP,128] author->patent accum
    float* accH = ws + 2 * NPf;             // [NA,128] patent->author accum
    float* p1   = ws + 2 * NPf + NAf;       // [NP,128]
    float* a1   = ws + 3 * NPf + NAf;       // [NA,128]
    float* cntC = ws + 3 * NPf + 2 * NAf;   // [NP]
    float* cntA = cntC + NP;                // [NP]
    float* cntH = cntA + NP;                // [NA]
    float* p2   = accH;                     // reuse: accH dead after layer-1 h

    const int B = 256;
    auto blk = [](long long n, int b) { return (int)((n + b - 1) / b); };

    // ---------------- layer 1 ----------------
    k_zero_f32<<<2048, B, 0, stream>>>(accC, 2 * NPf + NAf);
    k_zero_f32<<<256, B, 0, stream>>>(cntC, (long long)(2 * NP + NA));

    k_scatter_accum<<<blk((long long)Ec * 32, B), B, 0, stream>>>(x_p, c_src, c_dst, accC, cntC, Ec);
    k_scatter_accum<<<blk((long long)Ea * 32, B), B, 0, stream>>>(x_a, ao_src, ao_dst, accA, cntA, Ea);
    k_scatter_accum<<<blk((long long)Eh * 32, B), B, 0, stream>>>(x_p, ha_src, ha_dst, accH, cntH, Eh);

    k_mean_div<<<blk(NPf, B), B, 0, stream>>>(accC, cntC, NPf);
    k_mean_div<<<blk(NPf, B), B, 0, stream>>>(accA, cntA, NPf);
    k_mean_div<<<blk(NAf, B), B, 0, stream>>>(accH, cntH, NAf);

    int gP = (NP + RPB - 1) / RPB, gA = (NA + RPB - 1) / RPB;
    k_sage_gemm<0, 0><<<gP, B, 0, stream>>>(accC, x_p, l1c_Wl, l1c_bl, l1c_Wr, p1, NP, 0.5f);
    k_sage_gemm<1, 1><<<gP, B, 0, stream>>>(accA, x_p, l1a_Wl, l1a_bl, l1a_Wr, p1, NP, 0.5f);
    k_sage_gemm<0, 1><<<gA, B, 0, stream>>>(accH, x_a, l1h_Wl, l1h_bl, l1h_Wr, a1, NA, 1.0f);

    // ---------------- layer 2 ----------------
    k_zero_f32<<<2048, B, 0, stream>>>(accC, 2 * NPf);
    k_zero_f32<<<256, B, 0, stream>>>(cntC, (long long)(2 * NP));

    k_scatter_accum<<<blk((long long)Ec * 32, B), B, 0, stream>>>(p1, c_src, c_dst, accC, cntC, Ec);
    k_scatter_accum<<<blk((long long)Ea * 32, B), B, 0, stream>>>(a1, ao_src, ao_dst, accA, cntA, Ea);

    k_mean_div<<<blk(NPf, B), B, 0, stream>>>(accC, cntC, NPf);
    k_mean_div<<<blk(NPf, B), B, 0, stream>>>(accA, cntA, NPf);

    k_sage_gemm<0, 0><<<gP, B, 0, stream>>>(accC, p1, l2c_Wl, l2c_bl, l2c_Wr, p2, NP, 0.5f);
    k_sage_gemm<1, 1><<<gP, B, 0, stream>>>(accA, p1, l2a_Wl, l2a_bl, l2a_Wr, p2, NP, 0.5f);

    // ---------------- head ----------------
    k_head<<<blk((long long)NP * 8, B), B, 0, stream>>>(p2, lin_W, lin_b, (float*)d_out, NP);
}
